// UniGINConv_69063074119890
// MI455X (gfx1250) — matrix operand order repaired and checked by reading, not compiled, not run on hardware
//
#include <hip/hip_runtime.h>

#define NUM_NODES 50000
#define IN_DIM    512
#define OUT_DIM   512
#define NNZ       150000
#define NUM_EDGES 10000

typedef __bf16 bf16_t;
typedef __bf16 v16bf __attribute__((ext_vector_type(16)));
typedef __bf16 v8bf  __attribute__((ext_vector_type(8)));
typedef __bf16 v4bf  __attribute__((ext_vector_type(4)));
typedef float  v8f   __attribute__((ext_vector_type(8)));

// ---------------- Stage 1: node -> edge segment sum (atomics into L2-resident 20MB buffer)
__global__ void k_edge_accum(const float* __restrict__ X, const int* __restrict__ V,
                             const int* __restrict__ E, float* __restrict__ edge_sum,
                             float* __restrict__ count) {
    const int i = blockIdx.x;
    const int v = V[i];
    const int e = E[i];
    const float* __restrict__ src = X + (size_t)v * IN_DIM;
    float* __restrict__ dst = edge_sum + (size_t)e * IN_DIM;
    const int t = threadIdx.x;
    atomicAdd(dst + t,       src[t]);
    atomicAdd(dst + t + 256, src[t + 256]);
    if (t == 0) atomicAdd(count + e, 1.0f);
}

// ---------------- Edge mean: edge_emb = edge_sum / max(count, 1)
__global__ void k_edge_norm(float* __restrict__ edge, const float* __restrict__ count) {
    const int idx = blockIdx.x * blockDim.x + threadIdx.x;   // over NUM_EDGES*IN_DIM
    const float c = count[idx >> 9];                         // IN_DIM == 512
    edge[idx] *= 1.0f / fmaxf(c, 1.0f);
}

// ---------------- Stage 2: edge -> node segment sum
__global__ void k_node_accum(const float* __restrict__ edge_emb, const int* __restrict__ V,
                             const int* __restrict__ E, float* __restrict__ node_msg) {
    const int i = blockIdx.x;
    const int v = V[i];
    const int e = E[i];
    const float* __restrict__ src = edge_emb + (size_t)e * IN_DIM;
    float* __restrict__ dst = node_msg + (size_t)v * IN_DIM;
    const int t = threadIdx.x;
    atomicAdd(dst + t,       src[t]);
    atomicAdd(dst + t + 256, src[t + 256]);
}

// ---------------- Fuse: h = (1+eps)*X + node_msg, converted to bf16 for WMMA
__global__ void k_fuse(const float* __restrict__ X, const float* __restrict__ node_msg,
                       const float* __restrict__ eps_p, bf16_t* __restrict__ Hb) {
    const int idx = (blockIdx.x * blockDim.x + threadIdx.x) * 4;
    const float s = 1.0f + eps_p[0];
    const float4 x = *(const float4*)(X + idx);
    const float4 m = *(const float4*)(node_msg + idx);
    v4bf o;
    o[0] = (bf16_t)(s * x.x + m.x);
    o[1] = (bf16_t)(s * x.y + m.y);
    o[2] = (bf16_t)(s * x.z + m.z);
    o[3] = (bf16_t)(s * x.w + m.w);
    *(v4bf*)(Hb + idx) = o;
}

// ---------------- W (OUT_DIM x IN_DIM, row-major) -> bf16
__global__ void k_wcvt(const float* __restrict__ W, bf16_t* __restrict__ Wb) {
    const int idx = (blockIdx.x * blockDim.x + threadIdx.x) * 4;
    const float4 w = *(const float4*)(W + idx);
    v4bf o;
    o[0] = (bf16_t)w.x; o[1] = (bf16_t)w.y; o[2] = (bf16_t)w.z; o[3] = (bf16_t)w.w;
    *(v4bf*)(Wb + idx) = o;
}

// ---------------- GEMM: out = Hb @ Wb^T + bias   (M=50000, N=512, K=512)
// Block tile: 64(M) x 128(N), 256 threads = 8 waves. Wave w owns N columns
// [blockIdx.x*128 + w*16, +16) and four 16-row M subtiles.
// A strip (64x512 bf16 = 64KB) staged in LDS via GLOBAL_LOAD_ASYNC_TO_LDS
// (ASYNCcnt path: no VGPR round-trip), XOR-16B-chunk swizzle kills bank
// conflicts on the fragment reads.
__global__ __launch_bounds__(256) void k_gemm(const bf16_t* __restrict__ Hb,
                                              const bf16_t* __restrict__ Wb,
                                              const float* __restrict__ bias,
                                              float* __restrict__ out) {
    __shared__ bf16_t As[64 * IN_DIM];     // 65536 bytes, single shared var -> LDS offset 0
    const int mbase = blockIdx.y * 64;
    const int nbase = blockIdx.x * 128;

    // Async cooperative A-strip load: 4096 x 16B chunks, 16 per thread.
    // The LDS operand is derived from the real &As[...] pointer (ptrtoint) so
    // As escapes -> the "memory" clobber keeps the later ds_loads alive; the
    // low 32 bits of a flat shared pointer are exactly the wave-relative LDS
    // byte offset the async op's VDST operand wants.
#pragma unroll
    for (int j = 0; j < 16; ++j) {
        const int c    = threadIdx.x + 256 * j;     // chunk id: 64 chunks/row
        const int row  = c >> 6;
        const int cc   = c & 63;
        const int grow = mbase + row;
        if (grow < NUM_NODES) {
            bf16_t* dst = As + row * IN_DIM + ((cc ^ (row & 15)) * 8);
            const unsigned lds_off = (unsigned)(size_t)dst;
            const unsigned long long gaddr =
                (unsigned long long)(const void*)(Hb + (size_t)grow * IN_DIM + cc * 8);
            asm volatile("global_load_async_to_lds_b128 %0, %1, off"
                         :: "v"(lds_off), "v"(gaddr) : "memory");
        }
    }
    asm volatile("s_wait_asynccnt 0x0" ::: "memory");
    __syncthreads();

    const int lane = threadIdx.x & 31;
    const int wave = threadIdx.x >> 5;
    const int r    = lane & 15;
    const int half = lane >> 4;
    const int ncol = nbase + wave * 16;

    // B fragment: lane (r, half) holds column N=ncol+r, K = k0+half*8..+7 then k0+16+half*8..+7
    const bf16_t* __restrict__ wrow = Wb + (size_t)(ncol + r) * IN_DIM + half * 8;

    v8f acc[4] = {v8f{}, v8f{}, v8f{}, v8f{}};

    // Register-double-buffered B: break the WAR recurrence so the global load
    // for step k0+32 is in flight while the WMMAs of step k0 execute.
    v16bf bfrag = __builtin_shufflevector(*(const v8bf*)(wrow), *(const v8bf*)(wrow + 16), 0, 1, 2, 3, 4, 5, 6, 7, 8, 9, 10, 11, 12, 13, 14, 15);

#pragma unroll
    for (int k0 = 0; k0 < IN_DIM; k0 += 32) {
        v16bf bnext = bfrag;
        if (k0 + 32 < IN_DIM) bnext = __builtin_shufflevector(*(const v8bf*)(wrow + k0 + 32), *(const v8bf*)(wrow + k0 + 48), 0, 1, 2, 3, 4, 5, 6, 7, 8, 9, 10, 11, 12, 13, 14, 15);

        // Hoist all 8 A-fragment LDS loads ahead of the 4 WMMAs so the
        // scheduler can retire them with partial s_wait_dscnt values.
        v8bf alo[4], ahi[4];
#pragma unroll
        for (int m = 0; m < 4; ++m) {
            // A fragment: lane holds row M=16m+r; lo chunk = K k0+half*8..+7,
            // hi chunk = K k0+16+half*8..+7 (ISA 16-bit A layout), de-swizzled.
            const int row  = 16 * m + r;
            const bf16_t* arow = As + row * IN_DIM;
            const int cl = ((k0 >> 3) + half)     ^ (row & 15);
            const int ch = ((k0 >> 3) + 2 + half) ^ (row & 15);
            alo[m] = *(const v8bf*)(arow + cl * 8);
            ahi[m] = *(const v8bf*)(arow + ch * 8);
        }
#pragma unroll
        for (int m = 0; m < 4; ++m) {
            const v16bf afrag = __builtin_shufflevector(
                alo[m], ahi[m], 0, 1, 2, 3, 4, 5, 6, 7, 8, 9, 10, 11, 12, 13, 14, 15);
            acc[m] = __builtin_amdgcn_wmma_f32_16x16x32_bf16(
                false, afrag, false, bfrag, (short)0, acc[m], false, false);
        }
        bfrag = bnext;
    }

    const float bn = bias[ncol + r];
#pragma unroll
    for (int m = 0; m < 4; ++m) {
#pragma unroll
        for (int i = 0; i < 8; ++i) {
            // C layout: VGPR i, lanes 0-15 -> M=i; lanes 16-31 -> M=8+i; N=lane&15
            const int grow = mbase + 16 * m + half * 8 + i;
            if (grow < NUM_NODES)
                out[(size_t)grow * OUT_DIM + ncol + r] = acc[m][i] + bn;
        }
    }
}

extern "C" void kernel_launch(void* const* d_in, const int* in_sizes, int n_in,
                              void* d_out, int out_size, void* d_ws, size_t ws_size,
                              hipStream_t stream) {
    const float* X   = (const float*)d_in[0];
    const int*   V   = (const int*)d_in[1];
    const int*   E   = (const int*)d_in[2];
    // d_in[3] = num_edges scalar (compile-time constant here)
    const float* W   = (const float*)d_in[4];
    const float* b   = (const float*)d_in[5];
    const float* eps = (const float*)d_in[6];
    float* out = (float*)d_out;

    char* ws = (char*)d_ws;
    size_t off = 0;
    float* edge_sum = (float*)(ws + off); off += (size_t)NUM_EDGES * IN_DIM * 4;  // 20.48 MB
    float* count    = (float*)(ws + off); off += (size_t)NUM_EDGES * 4;
    off = (off + 255) & ~(size_t)255;
    float*  node_msg = (float*)(ws + off); off += (size_t)NUM_NODES * IN_DIM * 4; // 102.4 MB
    bf16_t* Hb       = (bf16_t*)(ws + off); off += (size_t)NUM_NODES * IN_DIM * 2; // 51.2 MB
    bf16_t* Wb       = (bf16_t*)(ws + off); off += (size_t)OUT_DIM * IN_DIM * 2;   // 0.5 MB

    hipMemsetAsync(edge_sum, 0, (size_t)NUM_EDGES * IN_DIM * 4, stream);
    hipMemsetAsync(count,    0, (size_t)NUM_EDGES * 4, stream);
    hipMemsetAsync(node_msg, 0, (size_t)NUM_NODES * IN_DIM * 4, stream);

    k_edge_accum<<<NNZ, 256, 0, stream>>>(X, V, E, edge_sum, count);
    k_edge_norm<<<(NUM_EDGES * IN_DIM) / 256, 256, 0, stream>>>(edge_sum, count);
    k_node_accum<<<NNZ, 256, 0, stream>>>(edge_sum, V, E, node_msg);
    k_fuse<<<(NUM_NODES * IN_DIM) / (256 * 4), 256, 0, stream>>>(X, node_msg, eps, Hb);
    k_wcvt<<<(OUT_DIM * IN_DIM) / (256 * 4), 256, 0, stream>>>(W, Wb);
    k_gemm<<<dim3(OUT_DIM / 128, (NUM_NODES + 63) / 64), 256, 0, stream>>>(Hb, Wb, b, out);
}
